// IPB_without_49177375539311
// MI455X (gfx1250) — matrix operand order repaired and checked by reading, not compiled, not run on hardware
//
#include <hip/hip_runtime.h>

typedef _Float16 half_t;
typedef __attribute__((ext_vector_type(16))) _Float16     v16h;
typedef __attribute__((ext_vector_type(8)))  _Float16     v8h;
typedef __attribute__((ext_vector_type(8)))  float        v8f;
typedef __attribute__((ext_vector_type(4)))  float        v4f;
typedef __attribute__((ext_vector_type(4)))  unsigned int v4u;
typedef __attribute__((ext_vector_type(8)))  int          v8i;
typedef __attribute__((ext_vector_type(4)))  int          v4i;

#define WAVES_PER_BLOCK 4
#define PW_BYTES 36864
// per-wave LDS region layout (bytes)
#define OFF_XO   0        // 64x28 f32 (7168B): x stage, later reused as out stage
#define OFF_Q    7168     // 64x32 f16 row-major (4096B), cols 28..31 zero
#define OFF_K    11264    // 64x32 f16 row-major
#define OFF_V    15360    // 32x64 f16 TRANSPOSED [d][tok], rows 28..31 zero
#define OFF_S    19456    // 64x64 f32 (16384B): raw sim -> attn (in place)
#define OFF_O1   7168     // 64x32 f32 (8192B): attn@v result (aliases Q+K)
#define OFF_SIGQ 35840    // 64 f32
#define OFF_SIGK 36096    // 64 f32
#define OFF_TH   36352    // 64 f32

__shared__ unsigned char smem[WAVES_PER_BLOCK * PW_BYTES];

#define WMMA_F16(A, B, C) \
  __builtin_amdgcn_wmma_f32_16x16x32_f16(false, (A), false, (B), (short)0, (C), false, false)

// ---- ISA 7.12.2 layout builders (vectorized: per-lane contiguous b128 runs) ----
// 16-bit A (16x32): lane m=l%16 reads K runs {off..off+7} and {off+16..off+23}, off=(l/16)*8
__device__ inline v16h loadA_f16_vec(const half_t* src, int row0, int ld, int lane) {
  int m = lane & 15; int off = (lane >> 4) * 8;
  const half_t* rp = src + (row0 + m) * ld + off;
  v8h lo = *(const v8h*)(rp);
  v8h hi = *(const v8h*)(rp + 16);
  v16h a;
#pragma unroll
  for (int t = 0; t < 8; ++t) { a[t] = lo[t]; a[t + 8] = hi[t]; }
  return a;
}
__device__ inline v16h loadA_f32_vec(const float* src, int row0, int ld, int k0, int kval, int lane) {
  int m = lane & 15; int off = (lane >> 4) * 8;
  const float* rp = src + (row0 + m) * ld + k0 + off;
  v4f a0 = *(const v4f*)(rp);
  v4f a1 = *(const v4f*)(rp + 4);
  v4f b0 = *(const v4f*)(rp + 16);
  v4f b1 = *(const v4f*)(rp + 20);
  v16h a;
#pragma unroll
  for (int t = 0; t < 4; ++t) {
    a[t]      = (off + t      < kval) ? (half_t)a0[t] : (half_t)0.f;
    a[t + 4]  = (off + 4 + t  < kval) ? (half_t)a1[t] : (half_t)0.f;
    a[t + 8]  = (off + 16 + t < kval) ? (half_t)b0[t] : (half_t)0.f;
    a[t + 12] = (off + 20 + t < kval) ? (half_t)b1[t] : (half_t)0.f;
  }
  return a;
}
// 16-bit B (32x16): lane n=l%16, K runs k0+{off..off+7} and k0+{off+16..off+23}, off=(l/16)*8; source row-major [N][K]
__device__ inline v16h loadB_cont_f16(const half_t* src, int ld, int n0, int k0, int lane) {
  const half_t* rp = src + (n0 + (lane & 15)) * ld + k0 + (lane >> 4) * 8;
  v8h lo = *(const v8h*)(rp);
  v8h hi = *(const v8h*)(rp + 16);
  v16h b;
#pragma unroll
  for (int t = 0; t < 8; ++t) { b[t] = lo[t]; b[t + 8] = hi[t]; }
  return b;
}
// weights from global: B[K][N] = W[(n0+N)*ldw + K], zero-padded
__device__ inline v16h loadB_w(const float* W, int ldw, int n0, int kval, int nval, int lane) {
  v16h b; int n = n0 + (lane & 15); int kbase = (lane >> 4) * 8;
#pragma unroll
  for (int t = 0; t < 16; ++t) {
    int K = (t < 8) ? (kbase + t) : (16 + kbase + (t - 8));
    b[t] = (K < kval && n < nval) ? (half_t)W[n * ldw + K] : (half_t)0.f;
  }
  return b;
}
// 32-bit C/D (16x16): VGPR r: M = m0 + r + 8*(l/16), N = n0 + l%16
__device__ inline void storeD_f16(half_t* dst, int ld, int m0, int n0, v8f c, int nval, int lane) {
  int ib = m0 + ((lane >> 4) << 3); int nn = n0 + (lane & 15);
  half_t z = (half_t)0.f;
#pragma unroll
  for (int r = 0; r < 8; ++r) dst[(ib + r) * ld + nn] = (nn < nval) ? (half_t)c[r] : z;
}
// transposed f16 D store: one contiguous v8h (= ds_store_b128) per tile per lane
__device__ inline void storeD_f16T(half_t* dstT, int ldT, int m0, int n0, v8f c, int nval, int lane) {
  int ib = m0 + ((lane >> 4) << 3); int nn = n0 + (lane & 15);
  v8h pk;
#pragma unroll
  for (int r = 0; r < 8; ++r) pk[r] = (nn < nval) ? (half_t)c[r] : (half_t)0.f;
  *(v8h*)(dstT + nn * ldT + ib) = pk;
}
__device__ inline void storeD_f32(float* dst, int ld, int m0, int n0, v8f c, int lane) {
  int ib = m0 + ((lane >> 4) << 3); int nn = n0 + (lane & 15);
#pragma unroll
  for (int r = 0; r < 8; ++r) dst[(ib + r) * ld + nn] = c[r];
}

// ---- Tensor Data Mover: 2D tile, dim0 = 224 f32 contiguous, dim1 = 8 rows, stride 7168
__device__ inline void make_tdm_desc(v4u& g0, v8i& g1, unsigned ldsOff, unsigned long long gaddr) {
  g0.x = 1u;                                   // count=1, user descriptor
  g0.y = ldsOff;                               // lds_addr
  g0.z = (unsigned)(gaddr & 0xFFFFFFFFu);      // global_addr[95:64]
  g0.w = (unsigned)((gaddr >> 32) & 0x1FFFFFFu) | (2u << 30);  // addr hi | type=2
  g1[0] = (2 << 16);    // data_size = 4B
  g1[1] = (224 << 16);  // tensor_dim0 = 224
  g1[2] = (8 << 16);    // tensor_dim1 = 8
  g1[3] = (224 << 16);  // tile_dim0 = 224
  g1[4] = 8;            // tile_dim1 = 8, tile_dim2 = 0
  g1[5] = 7168;         // tensor_dim0_stride = 256*28
  g1[6] = 0;
  g1[7] = 0;
}
__device__ inline void tdm_load(unsigned ldsOff, unsigned long long gaddr) {
  v4u g0; v8i g1; make_tdm_desc(g0, g1, ldsOff, gaddr);
  v4i g2 = {0, 0, 0, 0}, g3 = {0, 0, 0, 0};
#if __clang_major__ >= 23
  v8i g4 = {0, 0, 0, 0, 0, 0, 0, 0};
  __builtin_amdgcn_tensor_load_to_lds(g0, g1, g2, g3, g4, 0);
#else
  __builtin_amdgcn_tensor_load_to_lds(g0, g1, g2, g3, 0);
#endif
}
__device__ inline void tdm_store(unsigned ldsOff, unsigned long long gaddr) {
  v4u g0; v8i g1; make_tdm_desc(g0, g1, ldsOff, gaddr);
  v4i g2 = {0, 0, 0, 0}, g3 = {0, 0, 0, 0};
#if __clang_major__ >= 23
  v8i g4 = {0, 0, 0, 0, 0, 0, 0, 0};
  __builtin_amdgcn_tensor_store_from_lds(g0, g1, g2, g3, g4, 0);
#else
  __builtin_amdgcn_tensor_store_from_lds(g0, g1, g2, g3, 0);
#endif
}

__global__ __launch_bounds__(128, 1) void fa_forward_kernel(
    const float* __restrict__ x,
    const float* __restrict__ Wqk, const float* __restrict__ Wv,
    const float* __restrict__ Wout, const float* __restrict__ bout,
    const float* __restrict__ Wpcq, const float* __restrict__ bpcq,
    const float* __restrict__ Wpck, const float* __restrict__ bpck,
    const float* __restrict__ Wm1, const float* __restrict__ Wm2a,
    const float* __restrict__ Wm2b,
    float* __restrict__ out) {
  const int lane = threadIdx.x & 31;
  const int wave = __builtin_amdgcn_readfirstlane((int)(threadIdx.x >> 5));
  const int blk  = blockIdx.x;          // 2048 blocks
  const int win  = blk >> 1;            // window id 0..1023
  const int wh = win >> 5, ww = win & 31;
  const int Bi = ((blk & 1) << 2) + wave;  // batch-slice 0..7

  unsigned char* base = smem + wave * PW_BYTES;
  float*  sXO   = (float*)(base + OFF_XO);
  half_t* sQ    = (half_t*)(base + OFF_Q);
  half_t* sK    = (half_t*)(base + OFF_K);
  half_t* sVT   = (half_t*)(base + OFF_V);   // [32 d][64 tok]
  float*  sS    = (float*)(base + OFF_S);
  float*  sO1   = (float*)(base + OFF_O1);
  float*  sSigQ = (float*)(base + OFF_SIGQ);
  float*  sSigK = (float*)(base + OFF_SIGK);
  float*  sTh   = (float*)(base + OFF_TH);

  // ---------------- TDM: stage window x -> LDS ([64 tok][28 c] f32) -------
  const unsigned long long elemOff =
      (unsigned long long)(Bi * 256 + wh * 8) * 7168ull + (unsigned long long)ww * 224ull;
  tdm_load((unsigned)(wave * PW_BYTES + OFF_XO),
           (unsigned long long)x + 4ull * elemOff);
  __builtin_amdgcn_s_wait_tensorcnt(0);
  asm volatile("" ::: "memory");

  // ---------------- q,k,v projections: 3 low-pressure passes ---------------
  // pass 0: q -> sQ row-major; pass 1: k -> sK row-major; pass 2: v -> sVT transposed
#pragma unroll 1
  for (int pass = 0; pass < 3; ++pass) {
    const float* Wsrc = (pass == 0) ? Wqk : (pass == 1) ? (Wqk + 28 * 28) : Wv;
    v16h Bw0 = loadB_w(Wsrc, 28, 0, 28, 28, lane);
    v16h Bw1 = loadB_w(Wsrc, 28, 16, 28, 28, lane);
#pragma unroll
    for (int mt = 0; mt < 4; ++mt) {
      v16h A = loadA_f32_vec(sXO, mt * 16, 28, 0, 28, lane);
      v8f z = {0.f, 0.f, 0.f, 0.f, 0.f, 0.f, 0.f, 0.f};
      v8f d0 = WMMA_F16(A, Bw0, z);
      v8f d1 = WMMA_F16(A, Bw1, z);
      if (pass == 0) {
        storeD_f16(sQ, 32, mt * 16, 0, d0, 28, lane);
        storeD_f16(sQ, 32, mt * 16, 16, d1, 28, lane);
      } else if (pass == 1) {
        storeD_f16(sK, 32, mt * 16, 0, d0, 28, lane);
        storeD_f16(sK, 32, mt * 16, 16, d1, 28, lane);
      } else {
        storeD_f16T(sVT, 64, mt * 16, 0, d0, 28, lane);
        storeD_f16T(sVT, 64, mt * 16, 16, d1, 28, lane);
      }
    }
  }

  // ---------------- learned sigma per token (2 rows per lane) --------------
  {
    float bq = bpcq[0], bk = bpck[0];
#pragma unroll
    for (int rr = 0; rr < 2; ++rr) {
      int i = lane + rr * 32;
      float aq = bq, ak = bk;
#pragma unroll
      for (int d = 0; d < 28; ++d) {
        aq += (float)sQ[i * 32 + d] * Wpcq[d];
        ak += (float)sK[i * 32 + d] * Wpck[d];
      }
      sSigQ[i] = aq; sSigK[i] = ak;
    }
  }

  // ---------------- sim = q @ k^T (raw, f32 into LDS) ----------------------
#pragma unroll
  for (int mt = 0; mt < 4; ++mt) {
    v16h A = loadA_f16_vec(sQ, mt * 16, 32, lane);
#pragma unroll
    for (int nt = 0; nt < 4; ++nt) {
      v16h Bt = loadB_cont_f16(sK, 32, nt * 16, 0, lane);  // k^T
      v8f z = {0.f, 0.f, 0.f, 0.f, 0.f, 0.f, 0.f, 0.f};
      v8f d = WMMA_F16(A, Bt, z);
      storeD_f32(sS, 64, mt * 16, nt * 16, d, lane);
    }
  }

  // ---------------- MLP threshold theta (full dot minus diagonal) ----------
#pragma unroll
  for (int rr = 0; rr < 2; ++rr) {
    int i = lane + rr * 32;
    const v4f* rowp = (const v4f*)(sS + i * 64);
    float th = 0.f;
#pragma unroll
    for (int jv = 0; jv < 16; ++jv) {
      v4f s4 = rowp[jv];
      v4f w4 = *(const v4f*)(Wm1 + jv * 4);
      th += s4[0] * w4[0] + s4[1] * w4[1] + s4[2] * w4[2] + s4[3] * w4[3];
    }
    th -= sS[i * 64 + i] * Wm1[i];   // remove diagonal term
    sTh[i] = th;
  }
  float part = 0.f;
#pragma unroll
  for (int rr = 0; rr < 2; ++rr) {
    int i = lane + rr * 32;
    const v4f* ap = (const v4f*)(Wm2a + i * 64);
    float t = 0.f;
#pragma unroll
    for (int jv = 0; jv < 16; ++jv) {
      v4f a4 = ap[jv];
      v4f t4 = *(const v4f*)(sTh + jv * 4);
      t += a4[0] * t4[0] + a4[1] * t4[1] + a4[2] * t4[2] + a4[3] * t4[3];
    }
    t = (t >= 0.f) ? t : 0.1f * t;   // LeakyReLU(0.1)
    part += Wm2b[i] * t;
  }
#pragma unroll
  for (int off = 16; off > 0; off >>= 1) part += __shfl_xor(part, off, 32);
  const float thetaF = part;

  // ---------------- Sigma rescale + masked softmax (vector rows) -----------
#pragma unroll 1
  for (int rr = 0; rr < 2; ++rr) {
    int i = lane + rr * 32;
    float sq = sSigQ[i];
    v4f* rowp = (v4f*)(sS + i * 64);
    v4f rows[16];
    float mx = -3.0e38f;
#pragma unroll
    for (int jv = 0; jv < 16; ++jv) {
      v4f s4 = rowp[jv];
      v4f g4 = *(const v4f*)(sSigK + jv * 4);
      s4 = s4 * (sq * g4);
      rows[jv] = s4;
      mx = fmaxf(fmaxf(fmaxf(mx, s4[0]), fmaxf(s4[1], s4[2])), s4[3]);
    }
    float sum = 0.f;
#pragma unroll
    for (int jv = 0; jv < 16; ++jv) {
      v4f e4;
#pragma unroll
      for (int u = 0; u < 4; ++u) {
        float s = rows[jv][u];
        float e = __expf(s - mx);
        sum += e;
        e4[u] = (s > thetaF) ? e : 0.f;
      }
      rows[jv] = e4;
    }
    float inv = 1.f / sum;
#pragma unroll
    for (int jv = 0; jv < 16; ++jv) rowp[jv] = rows[jv] * inv;
  }

  // ---------------- out1 = attn @ v (K=64 -> two f16 K-steps) --------------
#pragma unroll
  for (int mt = 0; mt < 4; ++mt) {
#pragma unroll
    for (int nt = 0; nt < 2; ++nt) {
      v8f acc = {0.f, 0.f, 0.f, 0.f, 0.f, 0.f, 0.f, 0.f};
#pragma unroll
      for (int kt = 0; kt < 2; ++kt) {
        v16h A  = loadA_f32_vec(sS, mt * 16, 64, kt * 32, 32, lane);
        v16h Bm = loadB_cont_f16(sVT, 64, nt * 16, kt * 32, lane);  // v via [d][tok]
        acc = WMMA_F16(A, Bm, acc);
      }
      storeD_f32(sO1, 32, mt * 16, nt * 16, acc, lane);
    }
  }

  // ---------------- out = out1 @ Wout^T + bout -> sXO ([64][28]) -----------
  {
    v16h Bo0 = loadB_w(Wout, 28, 0, 28, 28, lane);
    v16h Bo1 = loadB_w(Wout, 28, 16, 28, 28, lane);
#pragma unroll
    for (int mt = 0; mt < 4; ++mt) {
      v16h A = loadA_f32_vec(sO1, mt * 16, 32, 0, 32, lane);  // cols 28..31 are zero
#pragma unroll
      for (int nt = 0; nt < 2; ++nt) {
        v8f z = {0.f, 0.f, 0.f, 0.f, 0.f, 0.f, 0.f, 0.f};
        v8f d = WMMA_F16(A, (nt == 0) ? Bo0 : Bo1, z);
        int nn = nt * 16 + (lane & 15);
        if (nn < 28) {
          float bo = bout[nn];
          int ib = mt * 16 + ((lane >> 4) << 3);
#pragma unroll
          for (int r = 0; r < 8; ++r) sXO[(ib + r) * 28 + nn] = d[r] + bo;
        }
      }
    }
  }

  // ---------------- TDM: store staged output back to global ----------------
  asm volatile("s_wait_dscnt 0" ::: "memory");
  tdm_store((unsigned)(wave * PW_BYTES + OFF_XO),
            (unsigned long long)out + 4ull * elemOff);
  __builtin_amdgcn_s_wait_tensorcnt(0);
}

extern "C" void kernel_launch(void* const* d_in, const int* in_sizes, int n_in,
                              void* d_out, int out_size, void* d_ws, size_t ws_size,
                              hipStream_t stream) {
  (void)in_sizes; (void)n_in; (void)out_size; (void)d_ws; (void)ws_size;
  const float* x    = (const float*)d_in[0];
  const float* Wqk  = (const float*)d_in[1];
  const float* Wv   = (const float*)d_in[2];
  const float* Wout = (const float*)d_in[3];
  const float* bout = (const float*)d_in[4];
  const float* Wpcq = (const float*)d_in[5];
  const float* bpcq = (const float*)d_in[6];
  const float* Wpck = (const float*)d_in[7];
  const float* bpck = (const float*)d_in[8];
  const float* Wm1  = (const float*)d_in[9];
  const float* Wm2a = (const float*)d_in[10];
  const float* Wm2b = (const float*)d_in[11];
  float* out = (float*)d_out;
  dim3 grid(2048), block(128);
  hipLaunchKernelGGL(fa_forward_kernel, grid, block, 0, stream,
                     x, Wqk, Wv, Wout, bout, Wpcq, bpcq, Wpck, bpck,
                     Wm1, Wm2a, Wm2b, out);
}
